// RNN2_55473797595757
// MI455X (gfx1250) — matrix operand order repaired and checked by reading, not compiled, not run on hardware
//
#include <hip/hip_runtime.h>

// ---------------------------------------------------------------------------
// Autoregressive LSTM forward for MI455X (gfx1250, wave32, WMMA + TDM).
// 143 sequential steps; per-step GEMMs in v_wmma_f32_16x16x32_f16 (f32 acc);
// LSTM cell fused into the gates GEMM via LDS gate exchange; activation (A)
// panels staged into LDS by the Tensor Data Mover (tensor_load_to_lds) with
// LDS padding for bank-conflict-free fragment reads; weights f16, L2-resident.
// ---------------------------------------------------------------------------

#define SEED_LEN   120
#define TARGET_LEN 24
#define NSTEP      (SEED_LEN + TARGET_LEN - 1)   // 143 steps
#define TT         (SEED_LEN + TARGET_LEN)       // 144 time slots in poses
#define B_SZ       512
#define POSE       135
#define LIN        256
#define RNN        1024
#define HEAD       960
#define HEADP      144                           // POSE padded to 16 for WMMA N

#define CHUNK_K    128                           // K halves per TDM chunk
#define A_LDS_LD   136                           // 128 + 8 halves pad (TDM pad)
#define NCHUNK     10                            // 2 (S,K=256) + 8 (H,K=1024)
#define ABUF_BYTES (64 * A_LDS_LD * 2)           // one A panel: 64 rows

typedef __attribute__((ext_vector_type(16))) _Float16 v16h;
typedef __attribute__((ext_vector_type(8)))  _Float16 v8h;
typedef __attribute__((ext_vector_type(8)))  float    v8f;
typedef __attribute__((ext_vector_type(4)))  unsigned v4u;
typedef __attribute__((ext_vector_type(8)))  int      v8i;
typedef __attribute__((ext_vector_type(4)))  int      v4i;

// ---- CDNA5 WMMA fragment helpers (contiguous-K order, same in A and B, wave32) --------------

// A-matrix 16x32 f16, row-major source with leading dim `ld` (halves).
// lanes 0-15: row M=lane, halves[0..15]=K 0..15
// lanes16-31: row M=lane-16, halves[0..15]=K 16..31
__device__ __forceinline__ v16h load_a_frag(const _Float16* base, int ld,
                                            int m0, int k0) {
  int lane = threadIdx.x & 31;
  const _Float16* p = base + (size_t)(m0 + (lane & 15)) * ld
                           + k0 + ((lane >> 4) << 4);
  v8h lo = *(const v8h*)p;
  v8h hi = *(const v8h*)(p + 8);
  v16h a;
#pragma unroll
  for (int i = 0; i < 8; ++i) { a[i] = lo[i]; a[i + 8] = hi[i]; }
  return a;
}

// B-matrix 32x16 f16 taken directly from a row-major [N,K] weight matrix:
// lane 0-15: column n=lane, 16 contiguous K halves at k0;
// lane16-31: column n=lane-16, 16 contiguous K halves at k0+16. One 32B load.
__device__ __forceinline__ v16h load_b_frag(const _Float16* w, int ld,
                                            int n0, int k0) {
  int lane = threadIdx.x & 31;
  return *(const v16h*)(w + (size_t)(n0 + (lane & 15)) * ld
                          + k0 + ((lane >> 4) << 4));
}

__device__ __forceinline__ v8f wmma16(v16h a, v16h b, v8f c) {
  // D = A x B + C, f16 inputs, f32 accumulate
  return __builtin_amdgcn_wmma_f32_16x16x32_f16(false, a, false, b,
                                                (short)0, c, false, false);
}

__device__ __forceinline__ float sigmoidf(float x) {
  return 1.0f / (1.0f + __expf(-x));
}

// ---- Tensor Data Mover: 2-D tile load, 64 rows x 128 halves ---------------
// D# per CDNA5 ISA: group0 = {count=1, lds_addr, global_addr,
// type=2}; group1 = {data_size=2B, pad_enable, pad_interval=5 (256B),
// pad_amount=3 (16B) -> LDS row pitch 136 halves, tile 128x64,
// tensor_dim0_stride = source row stride}. Huge tensor dims => no OOB zeroing
// (tiles are always interior). Groups 2/3 + extra group zero (2-D tensor).
// This toolchain exposes the 6-arg builtin:
//   (uint32x4 g0, int32x8 g1, int32x4, int32x4, int32x8, i32 cpol)
__device__ __forceinline__ void tdm_load_a64(const _Float16* gsrc,
                                             int row_stride_halves,
                                             unsigned lds_addr) {
  unsigned long long ga = (unsigned long long)(uintptr_t)gsrc;
  v4u g0;
  g0[0] = 1u;                                            // count=1, user mode
  g0[1] = lds_addr;                                      // LDS byte address
  g0[2] = (unsigned)ga;                                  // global addr [31:0]
  g0[3] = (unsigned)((ga >> 32) & 0x01FFFFFFu)           // global addr [56:32]
        | (2u << 30);                                    // type = 2 ("image")
  v8i g1;
  g1[0] = (int)((1u << 16)                               // data_size = 2 bytes
              | (1u << 20)                               // pad_enable
              | (5u << 22)                               // pad_interval: 256B
              | (3u << 25));                             // pad_amount: 4 dwords
  g1[1] = 0;                                             // abar=0, tdim0 lo16=0
  g1[2] = 0x4000;                                        // tdim0=1<<30, tdim1 lo
  g1[3] = (int)((128u << 16) | 0x4000u);                 // tdim1=1<<30, tile0=128
  g1[4] = 64;                                            // tile1=64, tile2=0
  g1[5] = row_stride_halves;                             // dim0 stride [31:0]
  g1[6] = 0;                                             // stride hi, dim1 stride
  g1[7] = 0;
  v4i z4 = {0, 0, 0, 0};
  v8i z8 = {0, 0, 0, 0, 0, 0, 0, 0};
  __builtin_amdgcn_tensor_load_to_lds(g0, g1, z4, z4, z8, 0);
}

// ---- one-time prep kernels -------------------------------------------------

__global__ void cvt_f16(const float* __restrict__ s, _Float16* __restrict__ d,
                        int n) {
  int i = blockIdx.x * 256 + threadIdx.x;
  if (i < n) d[i] = (_Float16)s[i];
}

__global__ void pack_p2(const float* __restrict__ p2w,
                        _Float16* __restrict__ d) {  // [HEADP][HEAD], zero pad
  int i = blockIdx.x * 256 + threadIdx.x;
  if (i < HEADP * HEAD)
    d[i] = (i < POSE * HEAD) ? (_Float16)p2w[i] : (_Float16)0.0f;
}

__global__ void prep_bias(const float* __restrict__ bih,
                          const float* __restrict__ bhh,
                          float* __restrict__ bg,
                          const float* __restrict__ p2b,
                          float* __restrict__ p2bp) {
  int i = blockIdx.x * 256 + threadIdx.x;
  if (i < 4 * RNN) bg[i] = bih[i] + bhh[i];
  if (i < HEADP)   p2bp[i] = (i < POSE) ? p2b[i] : 0.0f;
}

__global__ void zero_u32(unsigned* __restrict__ p, int n) {
  int i = blockIdx.x * 256 + threadIdx.x;
  if (i < n) p[i] = 0u;
}

// ---- per-step kernels ------------------------------------------------------

// s = inp @ lin_w^T + lin_b  (K=135, tiny -> f32 VALU), output f16 [B,LIN]
__global__ void lin_kernel(const float* __restrict__ poses,
                           const float* __restrict__ prev,
                           const float* __restrict__ lw,
                           const float* __restrict__ lb,
                           _Float16* __restrict__ S, int t) {
  __shared__ float xs[POSE];
  int b = blockIdx.x;
  int n = threadIdx.x;                 // 256 threads = LIN outputs
  if (n < POSE)
    xs[n] = (t < SEED_LEN) ? poses[((size_t)b * TT + t) * POSE + n]
                           : prev[(size_t)b * POSE + n];
  __syncthreads();
  float acc = lb[n];
  const float* wr = lw + (size_t)n * POSE;
#pragma unroll 5
  for (int k = 0; k < POSE; ++k) acc = fmaf(xs[k], wr[k], acc);
  S[(size_t)b * LIN + n] = (_Float16)acc;
}

// gates = S @ w_ih^T + Hin @ w_hh^T + bg, fused LSTM cell update.
// Block = 128 thr = 4 waves; wave g owns gate g's 64x32 tile at (m0,n0).
// A panel (64 rows of [S|Hin]) staged in LDS by the TDM, double-buffered,
// issued by wave 0, synced with s_wait_tensorcnt + workgroup barrier.
// Gate tiles exchanged through (reused) LDS, then c/h updated in-kernel.
__global__ void gates_kernel(const _Float16* __restrict__ S,
                             const _Float16* __restrict__ Hin,
                             const _Float16* __restrict__ Wih,
                             const _Float16* __restrict__ Whh,
                             const float*    __restrict__ bg,
                             float*          __restrict__ Cst,
                             _Float16*       __restrict__ Hout) {
  // A double buffer (2 x 17408B) and the 4x64x33 f32 gate-exchange scratch
  // (33792B) are never live at the same time -> share the same LDS bytes.
  __shared__ __align__(16) char smem[2 * ABUF_BYTES];
  float (*gt)[64][33] = (float (*)[64][33])smem;

  int g  = threadIdx.x >> 5;                      // wave id = gate (i,f,g,o)
  int m0 = blockIdx.x * 64;
  int n0 = blockIdx.y * 32;
  int nrow = g * RNN + n0;                        // absolute weight row base
  bool w0 = (threadIdx.x < 32);                   // TDM issuing wave

  if (w0) {
    tdm_load_a64(S + (size_t)m0 * LIN, LIN, (unsigned)(uintptr_t)smem);
    __builtin_amdgcn_s_wait_tensorcnt((short)0);
  }
  __syncthreads();

  v8f c[4][2] = {};
  for (int ci = 0; ci < NCHUNK; ++ci) {
    int nci = ci + 1;
    if (w0 && nci < NCHUNK) {                     // prefetch next A panel
      const _Float16* gp = (nci < 2)
          ? S   + (size_t)m0 * LIN + (size_t)nci * CHUNK_K
          : Hin + (size_t)m0 * RNN + (size_t)(nci - 2) * CHUNK_K;
      tdm_load_a64(gp, (nci < 2) ? LIN : RNN,
                   (unsigned)(uintptr_t)(smem + (nci & 1) * ABUF_BYTES));
    }
    const _Float16* As = (const _Float16*)(smem + (ci & 1) * ABUF_BYTES);
    const _Float16* Wp = (ci < 2) ? Wih : Whh;
    int ldw   = (ci < 2) ? LIN : RNN;
    int kbase = ((ci < 2) ? ci : (ci - 2)) * CHUNK_K;
#pragma unroll
    for (int j = 0; j < CHUNK_K / 32; ++j) {
      int kk = kbase + j * 32;
      v16h b0 = load_b_frag(Wp, ldw, nrow,      kk);
      v16h b1 = load_b_frag(Wp, ldw, nrow + 16, kk);
#pragma unroll
      for (int mi = 0; mi < 4; ++mi) {
        v16h a = load_a_frag(As, A_LDS_LD, mi * 16, j * 32);
        c[mi][0] = wmma16(a, b0, c[mi][0]);
        c[mi][1] = wmma16(a, b1, c[mi][1]);
      }
    }
    if (w0) __builtin_amdgcn_s_wait_tensorcnt((short)0);
    __syncthreads();                              // panel ready / consumed
  }

  // exchange gate tiles through LDS (A panels dead now), add biases
  int lane = threadIdx.x & 31;
  int ncol = lane & 15;
  int mrow = (lane >> 4) << 3;
  float bias0 = bg[nrow + ncol];
  float bias1 = bg[nrow + 16 + ncol];
#pragma unroll
  for (int mi = 0; mi < 4; ++mi)
#pragma unroll
    for (int r = 0; r < 8; ++r) {
      gt[g][16 * mi + mrow + r][ncol]      = c[mi][0][r] + bias0;
      gt[g][16 * mi + mrow + r][ncol + 16] = c[mi][1][r] + bias1;
    }
  __syncthreads();

  // fused LSTM cell: 128 threads x 16 elems = 64x32 tile
#pragma unroll
  for (int e = 0; e < 16; ++e) {
    int idx = (int)threadIdx.x + 128 * e;
    int r = idx >> 5, col = idx & 31;
    float iv = gt[0][r][col], fv = gt[1][r][col];
    float gv = gt[2][r][col], ov = gt[3][r][col];
    size_t gi = (size_t)(m0 + r) * RNN + (n0 + col);
    float cn = sigmoidf(fv) * Cst[gi] + sigmoidf(iv) * tanhf(gv);
    Cst[gi]  = cn;
    Hout[gi] = (_Float16)(sigmoidf(ov) * tanhf(cn));
  }
}

// a1 = relu(H @ p1^T + b1) -> f16 [B, HEAD]
__global__ void head1_kernel(const _Float16* __restrict__ H,
                             const _Float16* __restrict__ p1,
                             const float*    __restrict__ b1,
                             _Float16*       __restrict__ A1) {
  int w  = threadIdx.x >> 5;
  int m0 = blockIdx.x * 32;
  int n0 = blockIdx.y * 64 + w * 16;
  v8f c0 = {}, c1 = {};
  for (int k = 0; k < RNN; k += 32) {
    v16h a0 = load_a_frag(H, RNN, m0,      k);
    v16h a1 = load_a_frag(H, RNN, m0 + 16, k);
    v16h bb = load_b_frag(p1, RNN, n0, k);
    c0 = wmma16(a0, bb, c0);
    c1 = wmma16(a1, bb, c1);
  }
  int lane = threadIdx.x & 31;
  int n  = n0 + (lane & 15);
  int mb = (lane >> 4) << 3;
  float bias = b1[n];
#pragma unroll
  for (int r = 0; r < 8; ++r) {
    A1[(size_t)(m0 + mb + r)      * HEAD + n] = (_Float16)fmaxf(c0[r] + bias, 0.0f);
    A1[(size_t)(m0 + 16 + mb + r) * HEAD + n] = (_Float16)fmaxf(c1[r] + bias, 0.0f);
  }
}

// out = inp + a1 @ p2^T + b2 ; writes all_outputs, predictions slice, prev.
__global__ void head2_kernel(const _Float16* __restrict__ A1,
                             const _Float16* __restrict__ p2,
                             const float*    __restrict__ b2p,
                             const float*    __restrict__ poses,
                             float*          __restrict__ prev,
                             float*          __restrict__ out, int t) {
  int m0 = blockIdx.x * 32;
  int n0 = blockIdx.y * 16;
  v8f c0 = {}, c1 = {};
  for (int k = 0; k < HEAD; k += 32) {
    v16h a0 = load_a_frag(A1, HEAD, m0,      k);
    v16h a1 = load_a_frag(A1, HEAD, m0 + 16, k);
    v16h bb = load_b_frag(p2, HEAD, n0, k);
    c0 = wmma16(a0, bb, c0);
    c1 = wmma16(a1, bb, c1);
  }
  int lane = threadIdx.x & 31;
  int n  = n0 + (lane & 15);
  int mb = (lane >> 4) << 3;
  float bias  = b2p[n];
  bool  valid = (n < POSE);
  float* out_pred = out;                                      // [B][24][POSE]
  float* out_all  = out + (size_t)B_SZ * TARGET_LEN * POSE;   // [B][143][POSE]
#pragma unroll
  for (int mi = 0; mi < 2; ++mi) {
    v8f cc = mi ? c1 : c0;
#pragma unroll
    for (int r = 0; r < 8; ++r) {
      int b = m0 + mi * 16 + mb + r;
      if (valid) {
        float inp = (t < SEED_LEN) ? poses[((size_t)b * TT + t) * POSE + n]
                                   : prev[(size_t)b * POSE + n];
        float o = inp + cc[r] + bias;
        out_all[((size_t)b * NSTEP + t) * POSE + n] = o;
        if (t >= SEED_LEN - 1)
          out_pred[((size_t)b * TARGET_LEN + (t - (SEED_LEN - 1))) * POSE + n] = o;
        prev[(size_t)b * POSE + n] = o;   // next-step autoregressive input
      }
    }
  }
}

// ---------------------------------------------------------------------------

extern "C" void kernel_launch(void* const* d_in, const int* in_sizes, int n_in,
                              void* d_out, int out_size, void* d_ws,
                              size_t ws_size, hipStream_t stream) {
  const float* poses = (const float*)d_in[0];
  const float* lin_w = (const float*)d_in[1];
  const float* lin_b = (const float*)d_in[2];
  const float* w_ih  = (const float*)d_in[3];
  const float* w_hh  = (const float*)d_in[4];
  const float* b_ih  = (const float*)d_in[5];
  const float* b_hh  = (const float*)d_in[6];
  const float* p1_w  = (const float*)d_in[7];
  const float* p1_b  = (const float*)d_in[8];
  const float* p2_w  = (const float*)d_in[9];
  const float* p2_b  = (const float*)d_in[10];
  float* out = (float*)d_out;

  // workspace carve-up (~18.5 MB), 256B aligned chunks
  char* ws = (char*)d_ws;
  size_t off = 0;
  auto take = [&](size_t bytes) {
    char* p = ws + off;
    off = (off + bytes + 255) & ~(size_t)255;
    return p;
  };
  _Float16* wih_h = (_Float16*)take((size_t)4 * RNN * LIN * 2);
  _Float16* whh_h = (_Float16*)take((size_t)4 * RNN * RNN * 2);
  _Float16* p1h   = (_Float16*)take((size_t)HEAD * RNN * 2);
  _Float16* p2h   = (_Float16*)take((size_t)HEADP * HEAD * 2);
  float*    bg    = (float*)   take((size_t)4 * RNN * 4);
  float*    p2bp  = (float*)   take((size_t)HEADP * 4);
  _Float16* S     = (_Float16*)take((size_t)B_SZ * LIN * 2);
  _Float16* H0    = (_Float16*)take((size_t)B_SZ * RNN * 2);
  _Float16* H1    = (_Float16*)take((size_t)B_SZ * RNN * 2);
  float*    Cst   = (float*)   take((size_t)B_SZ * RNN * 4);
  _Float16* A1    = (_Float16*)take((size_t)B_SZ * HEAD * 2);
  float*    prev  = (float*)   take((size_t)B_SZ * POSE * 4);

  auto g1 = [](int n) { return dim3((unsigned)((n + 255) / 256)); };

  // one-time prep (runs every launch -> deterministic, no cached state)
  cvt_f16<<<g1(4 * RNN * LIN), 256, 0, stream>>>(w_ih, wih_h, 4 * RNN * LIN);
  cvt_f16<<<g1(4 * RNN * RNN), 256, 0, stream>>>(w_hh, whh_h, 4 * RNN * RNN);
  cvt_f16<<<g1(HEAD * RNN),   256, 0, stream>>>(p1_w, p1h, HEAD * RNN);
  pack_p2<<<g1(HEADP * HEAD), 256, 0, stream>>>(p2_w, p2h);
  prep_bias<<<g1(4 * RNN),    256, 0, stream>>>(b_ih, b_hh, bg, p2_b, p2bp);
  zero_u32<<<g1(B_SZ * RNN / 2), 256, 0, stream>>>((unsigned*)H0, B_SZ * RNN / 2);
  zero_u32<<<g1(B_SZ * RNN),     256, 0, stream>>>((unsigned*)Cst, B_SZ * RNN);

  _Float16* Hb[2] = {H0, H1};
  for (int t = 0; t < NSTEP; ++t) {
    _Float16* Hin  = Hb[t & 1];
    _Float16* Hout = Hb[(t + 1) & 1];
    lin_kernel<<<B_SZ, LIN, 0, stream>>>(poses, prev, lin_w, lin_b, S, t);
    gates_kernel<<<dim3(B_SZ / 64, RNN / 32), 128, 0, stream>>>(
        S, Hin, wih_h, whh_h, bg, Cst, Hout);
    head1_kernel<<<dim3(B_SZ / 32, HEAD / 64), 128, 0, stream>>>(
        Hout, p1h, p1_b, A1);
    head2_kernel<<<dim3(B_SZ / 32, HEADP / 16), 32, 0, stream>>>(
        A1, p2h, p2bp, poses, prev, out, t);
  }

  (void)in_sizes; (void)n_in; (void)out_size; (void)ws_size;
}
